// MHRB_53223234732092
// MI455X (gfx1250) — matrix operand order repaired and checked by reading, not compiled, not run on hardware
//
#include <hip/hip_runtime.h>

typedef _Float16 h16;
typedef __attribute__((ext_vector_type(16))) _Float16 v16h;
typedef __attribute__((ext_vector_type(8)))  _Float16 v8h;
typedef __attribute__((ext_vector_type(8)))  float    v8f;

#define CC 64
#define TT 32
#define HH 64
#define WW 64
#define BB 2
#define CH_STRIDE (TT*HH*WW)   /* 131072 */
#define T_STRIDE  (HH*WW)      /* 4096   */
#define COLS (TT*16)           /* 512 columns per workgroup */
#define ROWP 72                /* padded halves per column row: 144B = 16B aligned, 36-dword bank stride */

/* LDS byte offsets */
#define OFF_W    0
#define OFF_BIAS (OFF_W    + 4*64*64*2)     /* 32768  */
#define OFF_X    (OFF_BIAS + 4*64*4)        /* 33792  */
#define OFF_Z    (OFF_X    + COLS*ROWP*2)   /* 107520 */
#define OFF_F    (OFF_Z    + COLS*ROWP*2)   /* 181248 */
#define LDS_BYTES (OFF_F   + COLS*ROWP*2)   /* 254976 */

/* CDNA5 hardware tanh (TRANS op) — one instruction instead of a libm expansion. */
static __device__ inline float fast_tanh(float x) {
#if __has_builtin(__builtin_amdgcn_tanhf)
    return __builtin_amdgcn_tanhf(x);
#else
    float y;
    asm volatile("v_tanh_f32 %0, %1\n\tv_nop\n\tv_nop" : "=v"(y) : "v"(x));
    return y;
#endif
}
/* sigmoid(x) = 0.5*tanh(x/2) + 0.5 : tanh + mul + fma, no division/exp */
static __device__ inline float fast_sigmoid(float x) {
    return __builtin_fmaf(0.5f, fast_tanh(0.5f * x), 0.5f);
}

__global__ __launch_bounds__(256)
void fused_mlp_fopool(const float* __restrict__ x,
                      const float* __restrict__ wz1, const float* __restrict__ bz1,
                      const float* __restrict__ wz2, const float* __restrict__ bz2,
                      const float* __restrict__ wf1, const float* __restrict__ bf1,
                      const float* __restrict__ wf2, const float* __restrict__ bf2,
                      float* __restrict__ out)
{
    extern __shared__ char smem[];
    h16*   Wl = (h16*)  (smem + OFF_W);     /* [4][64][64] f16 row-major (o,c) */
    float* Bl = (float*)(smem + OFF_BIAS);  /* [4][64] */
    h16*   Xl = (h16*)  (smem + OFF_X);     /* [COLS][ROWP] col-major-in-c     */
    h16*   Zl = (h16*)  (smem + OFF_Z);     /* [COLS][ROWP] */
    h16*   Fl = (h16*)  (smem + OFF_F);     /* [COLS][ROWP] */

    const int tid = threadIdx.x;
    const int gid = blockIdx.x;
    const int wt  = gid & 3;          /* w tile (W/16 = 4) */
    const int hh  = (gid >> 2) & 63;  /* h row            */
    const int b   = gid >> 8;         /* batch            */
    const int w0  = wt * 16;

    /* ---- stage weights f32 -> f16 into LDS: [0]=wz1 [1]=wz2 [2]=wf1 [3]=wf2 ---- */
    for (int i = tid; i < 4*64*64; i += 256) {
        const int m = i >> 12, r = i & 4095;
        const float* src = (m == 0) ? wz1 : (m == 1) ? wz2 : (m == 2) ? wf1 : wf2;
        Wl[i] = (h16)src[r];
    }
    {   /* biases: 4*64 = 256 values, one per thread */
        const int m = tid >> 6, r = tid & 63;
        const float* src = (m == 0) ? bz1 : (m == 1) ? bz2 : (m == 2) ? bf1 : bf2;
        Bl[tid] = src[r];
    }

    /* ---- stage input tile f32 -> f16, coalesced reads, [col][c] layout ---- */
    const size_t base_in = (size_t)b * CC * CH_STRIDE + (size_t)hh * WW + w0;
    for (int i = tid; i < CC * COLS; i += 256) {
        const int p = i & 15, t = (i >> 4) & 31, c = i >> 9;
        const float v = x[base_in + (size_t)c * CH_STRIDE + (size_t)t * T_STRIDE + p];
        Xl[(t * 16 + p) * ROWP + c] = (h16)v;
    }
    __syncthreads();

    const int wave = tid >> 5;
    const int lane = tid & 31;
    const int hl   = lane >> 4;   /* lane half (0/1) */
    const int n    = lane & 15;   /* pixel column / matrix row within tile */

    for (int tt = 0; tt < 4; ++tt) {
        const int t    = wave * 4 + tt;   /* each wave owns 4 time steps */
        const int colb = t * 16;

        /* B fragments of X (K = 64 over two K-steps), in this kernel's contiguous k order (same as A):
           lane (hl,n), element e -> X[k = 32s + 16*hl + e][pixel n]  */
        v16h xf[2];
#pragma unroll
        for (int s = 0; s < 2; ++s) {
            const h16* px = &Xl[(colb + n) * ROWP + 32 * s + 16 * hl];
            v8h lo = *(const v8h*)px;
            v8h hi = *(const v8h*)(px + 8);
            xf[s] = __builtin_shufflevector(lo, hi, 0,1,2,3,4,5,6,7,8,9,10,11,12,13,14,15);
        }

#pragma unroll
        for (int mlp = 0; mlp < 2; ++mlp) {
            const h16*   W1 = Wl + (2 * mlp + 0) * 4096;
            const h16*   W2 = Wl + (2 * mlp + 1) * 4096;
            const float* B1 = Bl + (2 * mlp + 0) * 64;
            const float* B2 = Bl + (2 * mlp + 1) * 64;
            h16* ZFo = mlp ? Fl : Zl;

            /* ---- layer 1: hidden = W1 * X ---- */
            v8f acc[4];
#pragma unroll
            for (int mt = 0; mt < 4; ++mt) {
                v8f a = {0.f,0.f,0.f,0.f,0.f,0.f,0.f,0.f};
#pragma unroll
                for (int s = 0; s < 2; ++s) {
                    /* A layout: lane row m=n, runs k in [32s+16hl,+16) */
                    const h16* pw = W1 + (16 * mt + n) * 64 + 32 * s + 16 * hl;
                    v8h lo = *(const v8h*)pw;
                    v8h hi = *(const v8h*)(pw + 8);
                    v16h af = __builtin_shufflevector(lo, hi, 0,1,2,3,4,5,6,7,8,9,10,11,12,13,14,15);
                    a = __builtin_amdgcn_wmma_f32_16x16x32_f16(false, af, false, xf[s],
                                                               (short)0, a, false, false);
                }
                acc[mt] = a;
            }

            /* bias + tanh (hardware v_tanh_f32);
               D layout: lane holds chan 16*mt + 8*hl + r, pixel n */
            float hv0[8], hv1[8], hv2[8], hv3[8];
#pragma unroll
            for (int r = 0; r < 8; ++r) {
                hv0[r] = fast_tanh(acc[0][r] + B1[ 0 + 8 * hl + r]);
                hv1[r] = fast_tanh(acc[1][r] + B1[16 + 8 * hl + r]);
                hv2[r] = fast_tanh(acc[2][r] + B1[32 + 8 * hl + r]);
                hv3[r] = fast_tanh(acc[3][r] + B1[48 + 8 * hl + r]);
            }

            /* Repack D layout -> B layout with a single cross-half exchange:
               consumer half h needs tile (2s+h): its own half's 8 values at e=8h+r,
               the partner half's 8 values (shfl_xor 16) at e=8(1-h)+r.            */
            v16h hf[2];
#pragma unroll
            for (int s = 0; s < 2; ++s) {
                v8h lo8, hi8;
#pragma unroll
                for (int r = 0; r < 8; ++r) {
                    float e0, e1, o0, o1;
                    if (s == 0) { e0 = hv0[r]; e1 = hv1[r]; } else { e0 = hv2[r]; e1 = hv3[r]; }
                    const float own  = hl ? e1 : e0;     /* hv[2s+hl]   */
                    const float sent = hl ? e0 : e1;     /* hv[2s+1-hl] */
                    const float got  = __shfl_xor(sent, 16, 32);
                    o0 = hl ? got : own;                 /* element r    */
                    o1 = hl ? own : got;                 /* element 8+r  */
                    lo8[r] = (h16)o0;
                    hi8[r] = (h16)o1;
                }
                hf[s] = __builtin_shufflevector(lo8, hi8, 0,1,2,3,4,5,6,7,8,9,10,11,12,13,14,15);
            }

            /* ---- layer 2 + output activation, pack f16, ds_store_b128 ---- */
#pragma unroll
            for (int mt = 0; mt < 4; ++mt) {
                v8f a = {0.f,0.f,0.f,0.f,0.f,0.f,0.f,0.f};
#pragma unroll
                for (int s = 0; s < 2; ++s) {
                    const h16* pw = W2 + (16 * mt + n) * 64 + 32 * s + 16 * hl;
                    v8h lo = *(const v8h*)pw;
                    v8h hi = *(const v8h*)(pw + 8);
                    v16h af = __builtin_shufflevector(lo, hi, 0,1,2,3,4,5,6,7,8,9,10,11,12,13,14,15);
                    a = __builtin_amdgcn_wmma_f32_16x16x32_f16(false, af, false, hf[s],
                                                               (short)0, a, false, false);
                }
                v8h pk;
#pragma unroll
                for (int r = 0; r < 8; ++r) {
                    const float v = a[r] + B2[16 * mt + 8 * hl + r];
                    pk[r] = (h16)(mlp ? fast_sigmoid(v) : fast_tanh(v));
                }
                *(v8h*)&ZFo[(colb + n) * ROWP + 16 * mt + 8 * hl] = pk;
            }
        }
    }
    __syncthreads();

    /* ---- fused bidirectional fo-pool; _bidir flips cancel on Z/F and select
           scan direction: c<32 forward over t, c>=32 backward over t.
           Lane map: p = tid&15 -> 16 contiguous w -> coalesced 64B stores. ---- */
    const int p  = tid & 15;
    const int cs = tid >> 4;  /* 0..15 */
    const size_t base_out = (size_t)b * CC * CH_STRIDE + (size_t)hh * WW + w0 + p;
#pragma unroll
    for (int j = 0; j < 4; ++j) {
        const int c = cs + 16 * j;
        const int fwd = (c < 32);
        float hacc = 0.f;
        for (int stepi = 0; stepi < TT; ++stepi) {
            const int t   = fwd ? stepi : (TT - 1 - stepi);
            const int row = (t * 16 + p) * ROWP + c;
            const float z = (float)Zl[row];
            const float f = (float)Fl[row];
            hacc = f * hacc + (1.f - f) * z;
            out[base_out + (size_t)c * CH_STRIDE + (size_t)t * T_STRIDE] = hacc;
        }
    }
}

extern "C" void kernel_launch(void* const* d_in, const int* in_sizes, int n_in,
                              void* d_out, int out_size, void* d_ws, size_t ws_size,
                              hipStream_t stream) {
    (void)in_sizes; (void)n_in; (void)out_size; (void)d_ws; (void)ws_size;
    const float* x   = (const float*)d_in[0];
    const float* wz1 = (const float*)d_in[1];
    const float* bz1 = (const float*)d_in[2];
    const float* wz2 = (const float*)d_in[3];
    const float* bz2 = (const float*)d_in[4];
    const float* wf1 = (const float*)d_in[5];
    const float* bf1 = (const float*)d_in[6];
    const float* wf2 = (const float*)d_in[7];
    const float* bf2 = (const float*)d_in[8];
    float* out = (float*)d_out;

    /* allow large dynamic LDS (CDNA5 WGP: up to 320KB) */
    (void)hipFuncSetAttribute((const void*)fused_mlp_fopool,
                              hipFuncAttributeMaxDynamicSharedMemorySize, LDS_BYTES);

    dim3 grid(BB * HH * (WW / 16));  /* 2*64*4 = 512 workgroups */
    dim3 block(256);
    fused_mlp_fopool<<<grid, block, LDS_BYTES, stream>>>(
        x, wz1, bz1, wz2, bz2, wf1, bf1, wf2, bf2, out);
}
